// Attention_283467842303
// MI455X (gfx1250) — matrix operand order repaired and checked by reading, not compiled, not run on hardware
//
#include <hip/hip_runtime.h>
#include <cstdint>

typedef __attribute__((ext_vector_type(16))) __bf16    v16bf;
typedef __attribute__((ext_vector_type(8)))  float     v8f;
typedef __attribute__((ext_vector_type(4)))  uint32_t  u32x4;

#define DEVINL __device__ __forceinline__

DEVINL uint16_t f32_to_bf16(float f) {
  union { float f; uint32_t u; } x; x.f = f;
  uint32_t u = x.u + 0x7FFFu + ((x.u >> 16) & 1u);   // round-to-nearest-even
  return (uint16_t)(u >> 16);
}

union FragBF { v16bf v; u32x4 u[2]; };

// A-matrix fragment (16x32 bf16): lane L<16 -> row L, K {0..7,16..23};
// lane L>=16 -> row L-16, K {8..15,24..31}. base points at (row0, k0), stride in elements.
DEVINL v16bf load_frag_a(const uint16_t* base, int stride, int lane) {
  const uint16_t* p = base + (lane & 15) * stride + ((lane < 16) ? 0 : 8);
  FragBF f;
  f.u[0] = *(const u32x4*)(p);
  f.u[1] = *(const u32x4*)(p + 16);
  return f.v;
}

// B-matrix fragment (32x16 bf16) from a tile stored column-major-by-N
// (i.e. tile[n][k] row-major): lane L -> column L%16, K {0..7,16..23} for L<16, {8..15,24..31} for L>=16.
DEVINL v16bf load_frag_b(const uint16_t* base, int stride, int lane) {
  const uint16_t* p = base + (lane & 15) * stride + ((lane < 16) ? 0 : 8);
  FragBF f;
  f.u[0] = *(const u32x4*)(p);
  f.u[1] = *(const u32x4*)(p + 16);
  return f.v;
}

DEVINL v8f wmma_bf16(v16bf a, v16bf b, v8f c) {
  return __builtin_amdgcn_wmma_f32_16x16x32_bf16(false, a, false, b, (short)0, c,
                                                 false, false);
}

// CDNA5 async global->LDS DMA (16B per lane), tracked by ASYNCcnt.
// The low 32 bits of a generic LDS pointer are the LDS byte address.
DEVINL void async_copy_b128(uint32_t lds_addr, const void* gaddr) {
  asm volatile("global_load_async_to_lds_b128 %0, %1, off"
               :: "v"(lds_addr), "v"((uint64_t)(uintptr_t)gaddr)
               : "memory");
}
DEVINL void wait_async0() {
  asm volatile("s_wait_asynccnt 0" ::: "memory");
}
DEVINL uint32_t lds_off(const void* p) { return (uint32_t)(uintptr_t)p; }

// ---------------------------------------------------------------------------
// Kernel 1: qkv = x @ w_qkv^T  (M=8192, N=2304, K=768), bf16 out into
// workspace layout [3][B][H][N][64].
// grid = (2304/128, 8192/128) = (18, 64), block = 256 (8 waves)
// ---------------------------------------------------------------------------
__global__ __launch_bounds__(256) void qkv_gemm(const float* __restrict__ x,
                                                const float* __restrict__ w,
                                                uint16_t* __restrict__ qkv) {
  __shared__ __align__(16) uint16_t At[128 * 40];
  __shared__ __align__(16) uint16_t Bt[128 * 40];
  const int lane = threadIdx.x & 31, wave = threadIdx.x >> 5;
  const int m0 = blockIdx.y * 128;
  const int n0 = blockIdx.x * 128;
  const int wm = (wave >> 2) * 64;
  const int wn = (wave & 3) * 32;

  v8f acc[4][2];
#pragma unroll
  for (int i = 0; i < 4; ++i)
#pragma unroll
    for (int j = 0; j < 2; ++j) acc[i][j] = {};

  for (int k0 = 0; k0 < 768; k0 += 32) {
    {  // cooperative load + fp32->bf16 convert: each thread 16 floats of A and B
      const int r = threadIdx.x >> 1;
      const int c = (threadIdx.x & 1) * 16;
      const float4* pa = (const float4*)(x + (size_t)(m0 + r) * 768 + k0 + c);
      const float4* pb = (const float4*)(w + (size_t)(n0 + r) * 768 + k0 + c);
      uint16_t* da = At + r * 40 + c;
      uint16_t* db = Bt + r * 40 + c;
      if (k0 + 32 < 768) {  // prefetch next K-step (global_prefetch_b8)
        __builtin_prefetch((const char*)(pa) + 128, 0, 0);
        __builtin_prefetch((const char*)(pb) + 128, 0, 0);
      }
#pragma unroll
      for (int i = 0; i < 4; ++i) {
        float4 va = pa[i], vb = pb[i];
        da[4 * i + 0] = f32_to_bf16(va.x); da[4 * i + 1] = f32_to_bf16(va.y);
        da[4 * i + 2] = f32_to_bf16(va.z); da[4 * i + 3] = f32_to_bf16(va.w);
        db[4 * i + 0] = f32_to_bf16(vb.x); db[4 * i + 1] = f32_to_bf16(vb.y);
        db[4 * i + 2] = f32_to_bf16(vb.z); db[4 * i + 3] = f32_to_bf16(vb.w);
      }
    }
    __syncthreads();
    v16bf af[4], bfr[2];
#pragma unroll
    for (int i = 0; i < 4; ++i)
      af[i] = load_frag_a(At + (wm + i * 16) * 40, 40, lane);
#pragma unroll
    for (int j = 0; j < 2; ++j)
      bfr[j] = load_frag_b(Bt + (wn + j * 16) * 40, 40, lane);
#pragma unroll
    for (int i = 0; i < 4; ++i)
#pragma unroll
      for (int j = 0; j < 2; ++j)
        acc[i][j] = wmma_bf16(af[i], bfr[j], acc[i][j]);
    __syncthreads();
  }

  // epilogue: scatter bf16 into [3][B=8][H=12][N=1024][64]
#pragma unroll
  for (int i = 0; i < 4; ++i)
#pragma unroll
    for (int j = 0; j < 2; ++j) {
      const int mBase = m0 + wm + i * 16;
      const int dBase = n0 + wn + j * 16;
#pragma unroll
      for (int r = 0; r < 8; ++r) {
        const int m = mBase + r + 8 * (lane >> 4);
        const int d = dBase + (lane & 15);
        const int three = d / 768;
        const int rem = d - three * 768;
        const int h = rem >> 6, dh = rem & 63;
        const int b = m >> 10, n = m & 1023;
        const size_t idx =
            (((((size_t)three * 8 + b) * 12 + h) * 1024) + n) * 64 + dh;
        qkv[idx] = f32_to_bf16(acc[i][j][r]);
      }
    }
}

// ---------------------------------------------------------------------------
// Kernel 2: flash attention. grid = (N/128, H, B) = (8,12,8), block = 256.
// Each wave owns 16 query rows; online softmax stats live per half-wave row.
// K tile is brought in with CDNA5 async global->LDS DMA (ASYNCcnt).
// ---------------------------------------------------------------------------
__global__ __launch_bounds__(256) void attn_fa(const uint16_t* __restrict__ qkv,
                                               uint16_t* __restrict__ aout) {
  constexpr int B = 8, H = 12, N = 1024;
  __shared__ __align__(16) uint16_t Kt[128 * 72];   // [key][d], padded
  __shared__ __align__(16) uint16_t Vt[64 * 136];   // transposed: [d][key]
  __shared__ __align__(16) uint16_t Pt[128 * 136];  // [qrow][key], wave-private rows

  const int lane = threadIdx.x & 31, wave = threadIdx.x >> 5;
  const int b = blockIdx.z, h = blockIdx.y, q0 = blockIdx.x * 128;
  const size_t headOff = ((size_t)b * H + h) * N * 64;
  const uint16_t* qp = qkv + headOff;
  const uint16_t* kp = qkv + (size_t)1 * B * H * N * 64 + headOff;
  const uint16_t* vp = qkv + (size_t)2 * B * H * N * 64 + headOff;

  // Q fragments for this wave's 16 rows (Dh=64 -> two K-chunks of 32)
  v16bf aq[2];
  {
    const uint16_t* qrow = qp + (size_t)(q0 + wave * 16) * 64;
    aq[0] = load_frag_a(qrow, 64, lane);
    aq[1] = load_frag_a(qrow + 32, 64, lane);
  }

  v8f o[4];
#pragma unroll
  for (int df = 0; df < 4; ++df) o[df] = {};
  float mrow[8], lrow[8];
#pragma unroll
  for (int r = 0; r < 8; ++r) { mrow[r] = -3.0e38f; lrow[r] = 0.0f; }

  for (int jt = 0; jt < 8; ++jt) {
    const int kt0 = jt * 128;
    {
      const int r = threadIdx.x >> 1;
      const int c = (threadIdx.x & 1) * 32;
      // K tile: pure bf16 DMA, no VGPR round trip (64B per thread).
      const uint16_t* srck = kp + (size_t)(kt0 + r) * 64 + c;
      const uint32_t dstk = lds_off(Kt + r * 72 + c);
#pragma unroll
      for (int i = 0; i < 4; ++i)
        async_copy_b128(dstk + i * 16, srck + i * 8);
      // V tile: needs transpose -> explicit load + b16 scatter into LDS.
      const uint32_t* srcv = (const uint32_t*)(vp + (size_t)(kt0 + r) * 64 + c);
      if (jt + 1 < 8) {
        __builtin_prefetch(kp + (size_t)(kt0 + 128 + r) * 64 + c, 0, 0);
        __builtin_prefetch(vp + (size_t)(kt0 + 128 + r) * 64 + c, 0, 0);
      }
#pragma unroll
      for (int i = 0; i < 16; ++i) {
        const uint32_t w2 = srcv[i];
        Vt[(c + 2 * i) * 136 + r] = (uint16_t)(w2 & 0xFFFFu);
        Vt[(c + 2 * i + 1) * 136 + r] = (uint16_t)(w2 >> 16);
      }
    }
    wait_async0();   // our async K-tile DMA
    __syncthreads(); // cross-wave visibility (compiler adds ds waits)

    // S = Q K^T  (16 x 128 per wave)
    v8f s[8];
#pragma unroll
    for (int nf = 0; nf < 8; ++nf) {
      v16bf bk0 = load_frag_b(Kt + nf * 16 * 72, 72, lane);
      v16bf bk1 = load_frag_b(Kt + nf * 16 * 72 + 32, 72, lane);
      v8f z = {};
      z = wmma_bf16(aq[0], bk0, z);
      s[nf] = wmma_bf16(aq[1], bk1, z);
    }

    // online softmax (rows live in 16-lane halves -> xor-shuffle reductions)
    float tmax[8];
#pragma unroll
    for (int r = 0; r < 8; ++r) tmax[r] = -3.0e38f;
#pragma unroll
    for (int nf = 0; nf < 8; ++nf)
#pragma unroll
      for (int r = 0; r < 8; ++r) {
        const float v = s[nf][r] * 0.125f;  // scale = Dh^-0.5
        s[nf][r] = v;
        tmax[r] = fmaxf(tmax[r], v);
      }
#pragma unroll
    for (int r = 0; r < 8; ++r)
#pragma unroll
      for (int off = 1; off < 16; off <<= 1)
        tmax[r] = fmaxf(tmax[r], __shfl_xor(tmax[r], off, 32));

    float fac[8], rs[8];
#pragma unroll
    for (int r = 0; r < 8; ++r) {
      const float mn = fmaxf(mrow[r], tmax[r]);
      fac[r] = __expf(mrow[r] - mn);
      mrow[r] = mn;
      rs[r] = 0.0f;
    }
#pragma unroll
    for (int nf = 0; nf < 8; ++nf)
#pragma unroll
      for (int r = 0; r < 8; ++r) {
        const float p = __expf(s[nf][r] - mrow[r]);
        s[nf][r] = p;
        rs[r] += p;
      }
#pragma unroll
    for (int r = 0; r < 8; ++r) {
#pragma unroll
      for (int off = 1; off < 16; off <<= 1) rs[r] += __shfl_xor(rs[r], off, 32);
      lrow[r] = lrow[r] * fac[r] + rs[r];
    }
#pragma unroll
    for (int df = 0; df < 4; ++df)
#pragma unroll
      for (int r = 0; r < 8; ++r) o[df][r] *= fac[r];

    // store P (bf16) to this wave's private LDS rows; DS ops are in-order per
    // wave, so no cross-wave barrier is needed before re-reading it.
    {
      uint16_t* prow = Pt + wave * 16 * 136;
#pragma unroll
      for (int nf = 0; nf < 8; ++nf)
#pragma unroll
        for (int r = 0; r < 8; ++r)
          prow[(r + 8 * (lane >> 4)) * 136 + nf * 16 + (lane & 15)] =
              f32_to_bf16(s[nf][r]);
    }

    // O += P V  (K-dim = 128 keys, 4 chunks of 32)
#pragma unroll
    for (int kc = 0; kc < 4; ++kc) {
      v16bf ap = load_frag_a(Pt + wave * 16 * 136 + kc * 32, 136, lane);
#pragma unroll
      for (int df = 0; df < 4; ++df) {
        v16bf bv = load_frag_b(Vt + df * 16 * 136 + kc * 32, 136, lane);
        o[df] = wmma_bf16(ap, bv, o[df]);
      }
    }
    __syncthreads();  // Kt/Vt reuse next tile
  }

  // normalize + store bf16 into [b][n][h*64+d]
#pragma unroll
  for (int df = 0; df < 4; ++df)
#pragma unroll
    for (int r = 0; r < 8; ++r) {
      const int m = q0 + wave * 16 + r + 8 * (lane >> 4);
      const int dcol = h * 64 + df * 16 + (lane & 15);
      const float val = o[df][r] / lrow[r];
      aout[((size_t)b * N + m) * 768 + dcol] = f32_to_bf16(val);
    }
}

// ---------------------------------------------------------------------------
// Kernel 3: out = attn_out(bf16) @ w_proj^T + b_proj  (M=8192, N=768, K=768)
// A tile (already bf16) is DMA'd with async global->LDS; B tile is fp32->bf16.
// grid = (6, 64), block = 256
// ---------------------------------------------------------------------------
__global__ __launch_bounds__(256) void proj_gemm(const uint16_t* __restrict__ a,
                                                 const float* __restrict__ w,
                                                 const float* __restrict__ bias,
                                                 float* __restrict__ out) {
  __shared__ __align__(16) uint16_t At[128 * 40];
  __shared__ __align__(16) uint16_t Bt[128 * 40];
  const int lane = threadIdx.x & 31, wave = threadIdx.x >> 5;
  const int m0 = blockIdx.y * 128;
  const int n0 = blockIdx.x * 128;
  const int wm = (wave >> 2) * 64;
  const int wn = (wave & 3) * 32;

  v8f acc[4][2];
#pragma unroll
  for (int i = 0; i < 4; ++i)
#pragma unroll
    for (int j = 0; j < 2; ++j) acc[i][j] = {};

  for (int k0 = 0; k0 < 768; k0 += 32) {
    {
      const int r = threadIdx.x >> 1;
      const int c = (threadIdx.x & 1) * 16;
      // A tile: bf16 -> async DMA straight to LDS (32B per thread).
      const uint16_t* sa = a + (size_t)(m0 + r) * 768 + k0 + c;
      const uint32_t da = lds_off(At + r * 40 + c);
      async_copy_b128(da, sa);
      async_copy_b128(da + 16, sa + 8);
      // B tile: fp32 load + convert.
      const float4* pb = (const float4*)(w + (size_t)(n0 + r) * 768 + k0 + c);
      uint16_t* db = Bt + r * 40 + c;
      if (k0 + 32 < 768) {
        __builtin_prefetch(sa + 384, 0, 0);  // A, 384 elements (12 K-steps) ahead
        __builtin_prefetch((const char*)(pb) + 128, 0, 0);
      }
#pragma unroll
      for (int i = 0; i < 4; ++i) {
        float4 v4 = pb[i];
        db[4 * i + 0] = f32_to_bf16(v4.x); db[4 * i + 1] = f32_to_bf16(v4.y);
        db[4 * i + 2] = f32_to_bf16(v4.z); db[4 * i + 3] = f32_to_bf16(v4.w);
      }
    }
    wait_async0();
    __syncthreads();
    v16bf af[4], bfr[2];
#pragma unroll
    for (int i = 0; i < 4; ++i)
      af[i] = load_frag_a(At + (wm + i * 16) * 40, 40, lane);
#pragma unroll
    for (int j = 0; j < 2; ++j)
      bfr[j] = load_frag_b(Bt + (wn + j * 16) * 40, 40, lane);
#pragma unroll
    for (int i = 0; i < 4; ++i)
#pragma unroll
      for (int j = 0; j < 2; ++j)
        acc[i][j] = wmma_bf16(af[i], bfr[j], acc[i][j]);
    __syncthreads();
  }

#pragma unroll
  for (int i = 0; i < 4; ++i)
#pragma unroll
    for (int j = 0; j < 2; ++j)
#pragma unroll
      for (int r = 0; r < 8; ++r) {
        const int m = m0 + wm + i * 16 + r + 8 * (lane >> 4);
        const int n = n0 + wn + j * 16 + (lane & 15);
        out[(size_t)m * 768 + n] = acc[i][j][r] + bias[n];
      }
}

extern "C" void kernel_launch(void* const* d_in, const int* in_sizes, int n_in,
                              void* d_out, int out_size, void* d_ws, size_t ws_size,
                              hipStream_t stream) {
  (void)in_sizes; (void)n_in; (void)out_size; (void)ws_size;
  const float* x      = (const float*)d_in[0];
  const float* w_qkv  = (const float*)d_in[1];
  const float* w_proj = (const float*)d_in[2];
  const float* b_proj = (const float*)d_in[3];
  float* out = (float*)d_out;

  uint16_t* qkv_ws  = (uint16_t*)d_ws;                          // 3*8*12*1024*64 bf16
  uint16_t* aout_ws = qkv_ws + (size_t)3 * 8 * 12 * 1024 * 64;  // 8*1024*768 bf16

  qkv_gemm<<<dim3(18, 64), 256, 0, stream>>>(x, w_qkv, qkv_ws);
  attn_fa<<<dim3(8, 12, 8), 256, 0, stream>>>(qkv_ws, aout_ws);
  proj_gemm<<<dim3(6, 64), 256, 0, stream>>>(aout_ws, w_proj, b_proj, out);
}
